// DeformConv2d_74285754351893
// MI455X (gfx1250) — matrix operand order repaired and checked by reading, not compiled, not run on hardware
//
#include <hip/hip_runtime.h>

// ---------------------------------------------------------------------------
// Deformable Conv2d for MI455X (gfx1250), fused im2col-in-LDS + bf16 WMMA GEMM
// B=8, CIN=COUT=128, H=W=HO=WO=64, 3x3, stride1, pad1, dil1, groups=1
// ---------------------------------------------------------------------------

typedef __attribute__((ext_vector_type(16))) __bf16          v16bf;
typedef __attribute__((ext_vector_type(8)))  float           v8f;
typedef __attribute__((ext_vector_type(8)))  unsigned short  u16x8;

#define Bn    8
#define CIN   128
#define Hh    64
#define Ww    64
#define COUTn 128
#define KTAPS 9
#define HOo   64
#define WOo   64
#define NPIX  (HOo * WOo)        // 4096
#define TILEN 64                 // output pixels per workgroup (one row)
#define PITCH 136                // LDS row pitch in bf16 elems (128 + 8 pad)

static __device__ __forceinline__ unsigned short f2bf(float f) {
  unsigned int u = __float_as_uint(f);
  u += 0x7FFFu + ((u >> 16) & 1u);      // round-to-nearest-even
  return (unsigned short)(u >> 16);
}
static __device__ __forceinline__ float bf2f(unsigned short h) {
  return __uint_as_float(((unsigned int)h) << 16);
}

// ---------------------------------------------------------------------------
// Kernel 1: x [B,CIN,H,W] f32  ->  xt [B,H,W,CIN] bf16  (tiled transpose)
// ---------------------------------------------------------------------------
__global__ __launch_bounds__(256) void k_xpose(const float* __restrict__ x,
                                               unsigned short* __restrict__ xt) {
  __shared__ float tile[32][33];
  const int b  = blockIdx.z;
  const int s0 = blockIdx.x * 32;   // spatial (h*W + w)
  const int c0 = blockIdx.y * 32;   // channel
  const int tx = threadIdx.x;       // 0..31
  const int ty = threadIdx.y;       // 0..7
  const float* xb = x + (size_t)b * CIN * NPIX;
#pragma unroll
  for (int i = 0; i < 32; i += 8)
    tile[ty + i][tx] = xb[(size_t)(c0 + ty + i) * NPIX + (s0 + tx)];
  __syncthreads();
  unsigned short* xo = xt + ((size_t)b * NPIX + s0) * CIN + c0;
#pragma unroll
  for (int i = 0; i < 32; i += 8)
    xo[(size_t)(ty + i) * CIN + tx] = f2bf(tile[tx][ty + i]);
}

// ---------------------------------------------------------------------------
// Kernel 2: pack weight [COUT,CIN,3,3] f32 into per-lane A-fragment order:
// wpk[ ((k*4+kk)*8 + mtile)*32*16 + lane*16 + j ]  (bf16)
// A 16x32 bf16 layout: lanes<16: K = j(+8 hi-half offset at j>=8 -> +16..23),
// lanes>=16 shift K by +8.  klocal = j + (j>=8 ? 8 : 0) + (lane>=16 ? 8 : 0)
// ---------------------------------------------------------------------------
__global__ __launch_bounds__(256) void k_wpack(const float* __restrict__ w,
                                               unsigned short* __restrict__ wpk) {
  const int idx   = blockIdx.x * 256 + threadIdx.x;     // 0 .. 147455 exactly
  const int j     = idx & 15;
  const int lane  = (idx >> 4) & 31;
  const int mtile = (idx >> 9) & 7;
  const int kk    = (idx >> 12) & 3;
  const int k     = idx >> 14;                          // 0..8
  const int klocal = j + ((j >= 8) ? 8 : 0) + ((lane >= 16) ? 8 : 0);
  const int c = kk * 32 + klocal;                       // input channel
  const int m = mtile * 16 + (lane & 15);               // output channel
  wpk[idx] = f2bf(w[((size_t)m * CIN + c) * KTAPS + k]);
}

// ---------------------------------------------------------------------------
// Kernel 3: fused deformable-im2col (LDS) + WMMA GEMM + bias
// grid: NTOT/64 = 512 blocks, 256 threads (8 wave32)
// ---------------------------------------------------------------------------
__global__ __launch_bounds__(256) void k_deform_gemm(
    const float* __restrict__ off,          // [B, 18, HO, WO]
    const unsigned short* __restrict__ xt,  // [B, H, W, CIN] bf16
    const unsigned short* __restrict__ wpk, // packed A fragments bf16
    const float* __restrict__ bias,         // [COUT]
    float* __restrict__ out) {              // [B, COUT, HO, WO]
  __shared__ unsigned short sval[2][TILEN * PITCH];   // 2 x 17 KB, double buffer

  const int t     = threadIdx.x;
  const int lane  = t & 31;
  const int wave  = t >> 5;                 // 0..7  -> M-tile
  const int nbase = blockIdx.x * TILEN;     // global pixel base (row aligned)

  // --- staging identity: 64 pixels x 4 channel-chunks of 32 ---
  const int n_loc  = t >> 2;                // 0..63 local pixel
  const int cchunk = (t & 3) * 32;          // channel chunk base
  const int n_g = nbase + n_loc;
  const int b   = n_g >> 12;                // pixel's batch
  const int pix = n_g & (NPIX - 1);
  const int ho  = pix >> 6;
  const int wo  = pix & 63;
  const float* offb = off + (size_t)b * (2 * KTAPS) * NPIX + pix;
  const unsigned short* xtb = xt + (size_t)b * NPIX * CIN;

  auto stage = [&](int k, int buf) {
    const float dy = offb[(2 * k)     * NPIX];
    const float dx = offb[(2 * k + 1) * NPIX];
    const float py = (float)(ho - 1 + (k / 3)) + dy;
    const float px = (float)(wo - 1 + (k % 3)) + dx;
    const float fy = floorf(py), fx = floorf(px);
    const int   y0 = (int)fy,    x0 = (int)fx;
    const float ly = py - fy,    lx = px - fx;
    const float w00 = (1.f - ly) * (1.f - lx);
    const float w01 = (1.f - ly) * lx;
    const float w10 = ly * (1.f - lx);
    const float w11 = ly * lx;
    const int y1 = y0 + 1, x1 = x0 + 1;
    const bool vy0 = (y0 >= 0) && (y0 < Hh), vy1 = (y1 >= 0) && (y1 < Hh);
    const bool vx0 = (x0 >= 0) && (x0 < Ww), vx1 = (x1 >= 0) && (x1 < Ww);
    const float m00 = (vy0 && vx0) ? w00 : 0.f;
    const float m01 = (vy0 && vx1) ? w01 : 0.f;
    const float m10 = (vy1 && vx0) ? w10 : 0.f;
    const float m11 = (vy1 && vx1) ? w11 : 0.f;
    const int cy0 = min(max(y0, 0), Hh - 1), cy1 = min(max(y1, 0), Hh - 1);
    const int cx0 = min(max(x0, 0), Ww - 1), cx1 = min(max(x1, 0), Ww - 1);
    const unsigned short* p00 = xtb + ((cy0 * Ww + cx0) * CIN) + cchunk;
    const unsigned short* p01 = xtb + ((cy0 * Ww + cx1) * CIN) + cchunk;
    const unsigned short* p10 = xtb + ((cy1 * Ww + cx0) * CIN) + cchunk;
    const unsigned short* p11 = xtb + ((cy1 * Ww + cx1) * CIN) + cchunk;
    unsigned short* dst = &sval[buf][n_loc * PITCH + cchunk];
#pragma unroll
    for (int i = 0; i < 4; i++) {                 // 4 x 8 channels
      const u16x8 a00 = *(const u16x8*)(p00 + 8 * i);
      const u16x8 a01 = *(const u16x8*)(p01 + 8 * i);
      const u16x8 a10 = *(const u16x8*)(p10 + 8 * i);
      const u16x8 a11 = *(const u16x8*)(p11 + 8 * i);
      u16x8 r;
#pragma unroll
      for (int j = 0; j < 8; j++) {
        const float v = m00 * bf2f(a00[j]) + m01 * bf2f(a01[j]) +
                        m10 * bf2f(a10[j]) + m11 * bf2f(a11[j]);
        r[j] = f2bf(v);
      }
      *(u16x8*)(dst + 8 * i) = r;                 // ds_store_b128
    }
  };

  // --- accumulators: wave owns M-tile `wave`, all 4 N-tiles ---
  const v8f vzero = {0.f, 0.f, 0.f, 0.f, 0.f, 0.f, 0.f, 0.f};
  v8f acc[4] = {vzero, vzero, vzero, vzero};

  union V16U { v16bf v; u16x8 h[2]; };

  stage(0, 0);
  __syncthreads();

  for (int k = 0; k < KTAPS; k++) {
    if (k + 1 < KTAPS) stage(k + 1, (k + 1) & 1);   // prefetch next tap
    const unsigned short* sv = sval[k & 1];
#pragma unroll
    for (int kk = 0; kk < 4; kk++) {
      // A fragment: one aligned 32B load per lane from packed weights (L2)
      const unsigned short* ap =
          wpk + ((size_t)(((k * 4 + kk) * 8 + wave) * 32 + lane) << 4);
      const v16bf a = *(const v16bf*)ap;
      // B fragment base channel for this lane (32x16 K-major layout)
      const int c0 = kk * 32 + ((lane >= 16) ? 8 : 0);
#pragma unroll
      for (int tn = 0; tn < 4; tn++) {
        const int n = tn * 16 + (lane & 15);
        const unsigned short* bp = sv + n * PITCH + c0;
        V16U bu;
        bu.h[0] = *(const u16x8*)bp;        // ds_load_b128
        bu.h[1] = *(const u16x8*)(bp + 16); // ds_load_b128
        acc[tn] = __builtin_amdgcn_wmma_f32_16x16x32_bf16(
            false, a, false, bu.v, (short)0, acc[tn], false, false);
      }
    }
    __syncthreads();
  }

  // --- epilogue: bias + store f32, C/D 16x16 layout -> [B,COUT,HO,WO] ---
  const int m0 = wave * 16 + ((lane >= 16) ? 8 : 0);
  float bi[8];
#pragma unroll
  for (int r = 0; r < 8; r++) bi[r] = bias[m0 + r];

  const int bb      = nbase >> 12;
  const int pixbase = nbase & (NPIX - 1);
  float* outb = out + (size_t)bb * COUTn * NPIX;
#pragma unroll
  for (int tn = 0; tn < 4; tn++) {
    const int pp = pixbase + tn * 16 + (lane & 15);
#pragma unroll
    for (int r = 0; r < 8; r++)
      outb[(size_t)(m0 + r) * NPIX + pp] = acc[tn][r] + bi[r];
  }
}

// ---------------------------------------------------------------------------
extern "C" void kernel_launch(void* const* d_in, const int* in_sizes, int n_in,
                              void* d_out, int out_size, void* d_ws, size_t ws_size,
                              hipStream_t stream) {
  const float* x      = (const float*)d_in[0];  // [8,128,64,64]
  const float* offset = (const float*)d_in[1];  // [8,18,64,64]
  const float* weight = (const float*)d_in[2];  // [128,128,3,3]
  const float* bias   = (const float*)d_in[3];  // [128]
  float* out = (float*)d_out;                   // [8,128,64,64]

  unsigned short* xt  = (unsigned short*)d_ws;            // 4,194,304 bf16 = 8 MiB
  unsigned short* wpk = xt + (size_t)Bn * NPIX * CIN;     // 147,456 bf16 = 288 KiB

  // 1) NHWC bf16 transpose of x
  k_xpose<<<dim3(NPIX / 32, CIN / 32, Bn), dim3(32, 8), 0, stream>>>(x, xt);
  // 2) pack weights into A-fragment lane order
  k_wpack<<<(KTAPS * 16384) / 256, 256, 0, stream>>>(weight, wpk);
  // 3) fused deformable im2col + WMMA GEMM + bias
  k_deform_gemm<<<(Bn * NPIX) / TILEN, 256, 0, stream>>>(offset, xt, wpk, bias, out);
}
